// EdgeConvEncoder_84146999263442
// MI455X (gfx1250) — matrix operand order repaired and checked by reading, not compiled, not run on hardware
//
#include <hip/hip_runtime.h>
#include <stdint.h>

// ---------------- problem constants (from reference) ----------------
#define NNODES 50000
#define NEDGES 800000
#define EDIM   64
#define HID    256
#define ODIM   128

typedef __bf16 bf16;
typedef __attribute__((ext_vector_type(16))) __bf16 v16bf;
typedef __attribute__((ext_vector_type(8)))  __bf16 v8bf;
typedef __attribute__((ext_vector_type(8)))  float  v8f;
typedef __attribute__((ext_vector_type(4)))  float  v4f;

union AFrag { v16bf v; v8bf h[2]; };

// Each wave owns MB=2 16-row M-blocks (32 edges) so every B fragment
// loaded from memory feeds 2 WMMAs -> halves weight-fragment traffic.
#define MB 2
#define ROWS_PER_WAVE (16 * MB)             // 32 edges per wave
#define EDGES_PER_WG  (8 * ROWS_PER_WAVE)   // 256 edges per 8-wave workgroup

// LDS row pitch (bf16 elements) for the per-wave 32x256 h1 staging tile.
// 264 = 256 + 8 keeps the 16 row-readers on distinct bank quartets.
#define HPITCH 264
#define LDS_BYTES (8 * ROWS_PER_WAVE * HPITCH * 2)   // 135168 B / workgroup

__device__ __forceinline__ float relu(float x) { return __builtin_fmaxf(x, 0.0f); }

// ---------------- fragment helpers ----------------
// A-matrix 16x32 bf16 layout (ISA 7.12.2): lane L holds row M = L&15;
// elements 0..7  -> K = kbase + 8*half + (0..7)
// elements 8..15 -> K = kbase + 16 + 8*half + (0..7),  half = L>>4.
__device__ __forceinline__ v16bf load_a_bf16(const bf16* rowPtr, int kbase, int half) {
  AFrag f;
  f.h[0] = *(const v8bf*)(rowPtr + kbase + 8 * half);
  f.h[1] = *(const v8bf*)(rowPtr + kbase + 16 + 8 * half);
  return f.v;
}

// Same A layout, sourced from f32 (the scatter-sum buffer), scaled by 1/deg.
__device__ __forceinline__ v16bf load_a_f32(const float* rowPtr, float scale, int kbase, int half) {
  const v4f* p0 = (const v4f*)(rowPtr + kbase + 8 * half);
  const v4f* p1 = (const v4f*)(rowPtr + kbase + 16 + 8 * half);
  v4f a0 = p0[0], a1 = p0[1], b0 = p1[0], b1 = p1[1];
  AFrag f;
#pragma unroll
  for (int i = 0; i < 4; ++i) {
    f.h[0][i]     = (bf16)(a0[i] * scale);
    f.h[0][4 + i] = (bf16)(a1[i] * scale);
    f.h[1][i]     = (bf16)(b0[i] * scale);
    f.h[1][4 + i] = (bf16)(b1[i] * scale);
  }
  return f.v;
}

// B fragments are pre-packed per (ntile, kblock): 32 lanes x 16 bf16 contiguous.
__device__ __forceinline__ v16bf load_b(const bf16* pack, int numKB, int nt, int kb, int lane) {
  return *(const v16bf*)(pack + ((((long)nt * numKB + kb) * 32 + lane) << 4));
}

__device__ __forceinline__ v8f wmma_bf16(v16bf a, v16bf b, v8f c) {
  return __builtin_amdgcn_wmma_f32_16x16x32_bf16(false, a, false, b, (short)0, c, false, false);
}

__device__ __forceinline__ void atomAddF(float* p, float v) {
  unsafeAtomicAdd(p, v);   // global_atomic_add_f32 (L2 RMW)
}

// ---------------- small utility kernels ----------------
__global__ void zero_words(uint32_t* p, long n) {
  long i = (long)blockIdx.x * blockDim.x + threadIdx.x;
  long stride = (long)gridDim.x * blockDim.x;
  for (; i < n; i += stride) p[i] = 0u;
}

__global__ void cvt_f32_bf16(const float* __restrict__ in, bf16* __restrict__ out, long n) {
  long i = (long)blockIdx.x * blockDim.x + threadIdx.x;
  long stride = (long)gridDim.x * blockDim.x;
  for (; i < n; i += stride) out[i] = (bf16)in[i];
}

// Pack W[K,N] (row-major f32) into B-fragment order:
// dst[((nt*numKB + kb)*32 + lane)*16 + e] = W[32*kb + (e&7) + 8*(lane>>4) + 16*(e>>3)][16*nt + (lane&15)]
__global__ void pack_w(const float* __restrict__ W, bf16* __restrict__ dst, int K, int N) {
  int i = blockIdx.x * blockDim.x + threadIdx.x;
  if (i >= K * N) return;
  int k = i / N, n = i % N;
  int kb = k >> 5, kin = k & 31, nt = n >> 4, nin = n & 15;
  int lane = ((kin >> 3) & 1) * 16 + nin;
  int e = (kin & 7) + ((kin >> 4) << 3);
  int numKB = K >> 5;
  dst[(((nt * numKB + kb) * 32 + lane) << 4) + e] = (bf16)W[i];
}

__global__ void degree_count(const int* __restrict__ dst, int* __restrict__ deg, int E) {
  int i = blockIdx.x * blockDim.x + threadIdx.x;
  if (i < E) atomicAdd(&deg[dst[i]], 1);
}

// ---------------- layer 1 edge MLP + scatter-add ----------------
// per wave: 32 edges (2 M-blocks). GEMM1: [32,64]x[64,256] -> relu -> LDS bf16
//           GEMM2: [32,256]x[256,256] -> relu -> atomic scatter into sums[dst]
__global__ void __launch_bounds__(256) edge_mlp1(
    const bf16* __restrict__ ea, const int* __restrict__ dst,
    const bf16* __restrict__ w1, const float* __restrict__ b1,
    const bf16* __restrict__ w2, const float* __restrict__ b2,
    float* __restrict__ sums)
{
  extern __shared__ bf16 lds[];
  const int lane = threadIdx.x & 31, wave = threadIdx.x >> 5;
  const int half = lane >> 4, ln = lane & 15;
  const long eb = ((long)blockIdx.x * 8 + wave) * ROWS_PER_WAVE;
  bf16* h = lds + wave * ROWS_PER_WAVE * HPITCH;

  const bf16* arow[MB];
#pragma unroll
  for (int m = 0; m < MB; ++m) arow[m] = ea + (eb + 16 * m + ln) * EDIM;

  // stage 1: K=64 (2 kblocks), N=256 (16 ntiles)
  for (int nt = 0; nt < 16; ++nt) {
    float bv = b1[nt * 16 + ln];
    v8f acc[MB];
#pragma unroll
    for (int m = 0; m < MB; ++m)
#pragma unroll
      for (int i = 0; i < 8; ++i) acc[m][i] = bv;
#pragma unroll
    for (int kb = 0; kb < 2; ++kb) {
      v16bf b = load_b(w1, 2, nt, kb, lane);
#pragma unroll
      for (int m = 0; m < MB; ++m)
        acc[m] = wmma_bf16(load_a_bf16(arow[m], kb * 32, half), b, acc[m]);
    }
#pragma unroll
    for (int m = 0; m < MB; ++m)
#pragma unroll
      for (int r = 0; r < 8; ++r)
        h[(16 * m + 8 * half + r) * HPITCH + nt * 16 + ln] = (bf16)relu(acc[m][r]);
  }

  // stage 2: K=256 (8 kblocks), N=256; same-wave LDS RAW handled by dscnt waits
  const bf16* hrow[MB];
#pragma unroll
  for (int m = 0; m < MB; ++m) hrow[m] = h + (16 * m + ln) * HPITCH;

  for (int nt = 0; nt < 16; ++nt) {
    float bv = b2[nt * 16 + ln];
    v8f acc[MB];
#pragma unroll
    for (int m = 0; m < MB; ++m)
#pragma unroll
      for (int i = 0; i < 8; ++i) acc[m][i] = bv;
#pragma unroll
    for (int kb = 0; kb < 8; ++kb) {
      v16bf b = load_b(w2, 8, nt, kb, lane);
#pragma unroll
      for (int m = 0; m < MB; ++m)
        acc[m] = wmma_bf16(load_a_bf16(hrow[m], kb * 32, half), b, acc[m]);
    }
#pragma unroll
    for (int m = 0; m < MB; ++m)
#pragma unroll
      for (int r = 0; r < 8; ++r) {
        long e = eb + 16 * m + 8 * half + r;
        atomAddF(&sums[(long)dst[e] * HID + nt * 16 + ln], relu(acc[m][r]));
      }
  }
}

// ---------------- layer 2 edge MLP (concat input via gathered A-frags) ----------------
// msg_in = [edge_attr(64) | x[src](256)]  => K=320 (10 kblocks): kb<2 from ea, kb>=2 from x[src]
__global__ void __launch_bounds__(256) edge_mlp2(
    const bf16* __restrict__ ea, const bf16* __restrict__ xsrc,
    const int* __restrict__ src, const int* __restrict__ dst,
    const bf16* __restrict__ w1, const float* __restrict__ b1,
    const bf16* __restrict__ w2, const float* __restrict__ b2,
    float* __restrict__ sums)
{
  extern __shared__ bf16 lds[];
  const int lane = threadIdx.x & 31, wave = threadIdx.x >> 5;
  const int half = lane >> 4, ln = lane & 15;
  const long eb = ((long)blockIdx.x * 8 + wave) * ROWS_PER_WAVE;
  bf16* h = lds + wave * ROWS_PER_WAVE * HPITCH;

  const bf16* arow_e[MB];
  const bf16* arow_x[MB];
#pragma unroll
  for (int m = 0; m < MB; ++m) {
    arow_e[m] = ea + (eb + 16 * m + ln) * EDIM;
    arow_x[m] = xsrc + (long)src[eb + 16 * m + ln] * HID;  // per-lane row gather
  }

  for (int nt = 0; nt < 16; ++nt) {
    float bv = b1[nt * 16 + ln];
    v8f acc[MB];
#pragma unroll
    for (int m = 0; m < MB; ++m)
#pragma unroll
      for (int i = 0; i < 8; ++i) acc[m][i] = bv;
#pragma unroll
    for (int kb = 0; kb < 10; ++kb) {
      v16bf b = load_b(w1, 10, nt, kb, lane);
#pragma unroll
      for (int m = 0; m < MB; ++m) {
        v16bf a = (kb < 2) ? load_a_bf16(arow_e[m], kb * 32, half)
                           : load_a_bf16(arow_x[m], (kb - 2) * 32, half);
        acc[m] = wmma_bf16(a, b, acc[m]);
      }
    }
#pragma unroll
    for (int m = 0; m < MB; ++m)
#pragma unroll
      for (int r = 0; r < 8; ++r)
        h[(16 * m + 8 * half + r) * HPITCH + nt * 16 + ln] = (bf16)relu(acc[m][r]);
  }

  const bf16* hrow[MB];
#pragma unroll
  for (int m = 0; m < MB; ++m) hrow[m] = h + (16 * m + ln) * HPITCH;

  for (int nt = 0; nt < 16; ++nt) {
    float bv = b2[nt * 16 + ln];
    v8f acc[MB];
#pragma unroll
    for (int m = 0; m < MB; ++m)
#pragma unroll
      for (int i = 0; i < 8; ++i) acc[m][i] = bv;
#pragma unroll
    for (int kb = 0; kb < 8; ++kb) {
      v16bf b = load_b(w2, 8, nt, kb, lane);
#pragma unroll
      for (int m = 0; m < MB; ++m)
        acc[m] = wmma_bf16(load_a_bf16(hrow[m], kb * 32, half), b, acc[m]);
    }
#pragma unroll
    for (int m = 0; m < MB; ++m)
#pragma unroll
      for (int r = 0; r < 8; ++r) {
        long e = eb + 16 * m + 8 * half + r;
        atomAddF(&sums[(long)dst[e] * HID + nt * 16 + ln], relu(acc[m][r]));
      }
  }
}

// ---------------- node update: relu((sums/deg) @ Wu + bu) ----------------
template <bool BF16OUT>
__global__ void __launch_bounds__(256) node_update(
    const float* __restrict__ sums, const int* __restrict__ deg,
    const bf16* __restrict__ wu, const float* __restrict__ bu,
    int ncols, bf16* __restrict__ outb, float* __restrict__ outf)
{
  const int lane = threadIdx.x & 31, wave = threadIdx.x >> 5;
  const int half = lane >> 4, ln = lane & 15;
  const long nb = (long)blockIdx.x * 8 + wave;
  if (nb >= NNODES / 16) return;                         // wave-uniform exit (EXEC all-1 for WMMA)
  const long row = nb * 16 + ln;
  const float* rowPtr = sums + row * HID;
  int d = deg[row];
  float scale = 1.0f / (float)(d > 0 ? d : 1);
  const int ntiles = ncols >> 4;

  for (int nt = 0; nt < ntiles; ++nt) {
    float bv = bu[nt * 16 + ln];
    v8f acc;
#pragma unroll
    for (int i = 0; i < 8; ++i) acc[i] = bv;
#pragma unroll
    for (int kb = 0; kb < 8; ++kb)                        // K = 256
      acc = wmma_bf16(load_a_f32(rowPtr, scale, kb * 32, half), load_b(wu, 8, nt, kb, lane), acc);
#pragma unroll
    for (int r = 0; r < 8; ++r) {
      float x = relu(acc[r]);
      long m = nb * 16 + 8 * half + r;
      if (BF16OUT) outb[m * ncols + nt * 16 + ln] = (bf16)x;
      else         outf[m * ncols + nt * 16 + ln] = x;
    }
  }
}

// ---------------- host side ----------------
static inline size_t alignup(size_t x) { return (x + 255) & ~(size_t)255; }

extern "C" void kernel_launch(void* const* d_in, const int* in_sizes, int n_in,
                              void* d_out, int out_size, void* d_ws, size_t ws_size,
                              hipStream_t stream) {
  (void)in_sizes; (void)n_in; (void)out_size; (void)ws_size;

  const int*   eidx = (const int*)d_in[0];      // [2,E] flat
  const float* ea32 = (const float*)d_in[1];    // [E,64]
  const float* W1a = (const float*)d_in[2];  const float* b1a = (const float*)d_in[3];
  const float* W2a = (const float*)d_in[4];  const float* b2a = (const float*)d_in[5];
  const float* Wua = (const float*)d_in[6];  const float* bua = (const float*)d_in[7];
  const float* W1b = (const float*)d_in[8];  const float* b1b = (const float*)d_in[9];
  const float* W2b = (const float*)d_in[10]; const float* b2b = (const float*)d_in[11];
  const float* Wub = (const float*)d_in[12]; const float* bub = (const float*)d_in[13];

  const int* src = eidx;
  const int* dst = eidx + NEDGES;

  // workspace carve-up
  char* ws = (char*)d_ws;
  bf16* wpack = (bf16*)ws;                                   // 327680 bf16 = 655360 B
  size_t o = alignup(327680 * sizeof(bf16));
  bf16* eab = (bf16*)(ws + o);  o = alignup(o + (size_t)NEDGES * EDIM * sizeof(bf16));
  float* sums = (float*)(ws + o); o = alignup(o + (size_t)NNODES * HID * sizeof(float));
  int* deg = (int*)(ws + o);      o = alignup(o + (size_t)NNODES * sizeof(int));
  bf16* xb = (bf16*)(ws + o);     // NNODES*HID bf16

  // packed-weight element offsets
  bf16* pW1a = wpack + 0;        // 64x256   -> 16384
  bf16* pW2a = wpack + 16384;    // 256x256  -> 65536
  bf16* pWua = wpack + 81920;    // 256x256
  bf16* pW1b = wpack + 147456;   // 320x256  -> 81920
  bf16* pW2b = wpack + 229376;   // 256x256
  bf16* pWub = wpack + 294912;   // 256x128  -> 32768

  // 1) pack weights to bf16 B-fragment layout
  pack_w<<<dim3((64 * 256 + 255) / 256), dim3(256), 0, stream>>>(W1a, pW1a, 64, 256);
  pack_w<<<dim3((256 * 256 + 255) / 256), dim3(256), 0, stream>>>(W2a, pW2a, 256, 256);
  pack_w<<<dim3((256 * 256 + 255) / 256), dim3(256), 0, stream>>>(Wua, pWua, 256, 256);
  pack_w<<<dim3((320 * 256 + 255) / 256), dim3(256), 0, stream>>>(W1b, pW1b, 320, 256);
  pack_w<<<dim3((256 * 256 + 255) / 256), dim3(256), 0, stream>>>(W2b, pW2b, 256, 256);
  pack_w<<<dim3((256 * 128 + 255) / 256), dim3(256), 0, stream>>>(Wub, pWub, 256, 128);

  // 2) edge_attr -> bf16 (reused by both layers)
  cvt_f32_bf16<<<dim3(2048), dim3(256), 0, stream>>>(ea32, eab, (long)NEDGES * EDIM);

  // 3) degrees (shared by both layers)
  zero_words<<<dim3(256), dim3(256), 0, stream>>>((uint32_t*)deg, (long)NNODES);
  degree_count<<<dim3((NEDGES + 255) / 256), dim3(256), 0, stream>>>(dst, deg, NEDGES);

  // 4) layer 1
  zero_words<<<dim3(2048), dim3(256), 0, stream>>>((uint32_t*)sums, (long)NNODES * HID);
  edge_mlp1<<<dim3(NEDGES / EDGES_PER_WG), dim3(256), LDS_BYTES, stream>>>(
      eab, dst, pW1a, b1a, pW2a, b2a, sums);
  node_update<true><<<dim3((NNODES / 16 + 7) / 8), dim3(256), 0, stream>>>(
      sums, deg, pWua, bua, HID, xb, nullptr);

  // 5) layer 2
  zero_words<<<dim3(2048), dim3(256), 0, stream>>>((uint32_t*)sums, (long)NNODES * HID);
  edge_mlp2<<<dim3(NEDGES / EDGES_PER_WG), dim3(256), LDS_BYTES, stream>>>(
      eab, xb, src, dst, pW1b, b1b, pW2b, b2b, sums);
  node_update<false><<<dim3((NNODES / 16 + 7) / 8), dim3(256), 0, stream>>>(
      sums, deg, pWub, bub, ODIM, nullptr, (float*)d_out);
}
